// Block_59983513256170
// MI455X (gfx1250) — matrix operand order repaired and checked by reading, not compiled, not run on hardware
//
#include <hip/hip_runtime.h>
#include <hip/hip_bf16.h>

// ---------------------------------------------------------------------------
// Types
// ---------------------------------------------------------------------------
typedef __attribute__((ext_vector_type(16))) _Float16 v16h;
typedef __attribute__((ext_vector_type(8)))  _Float16 v8h;
typedef __attribute__((ext_vector_type(4)))  _Float16 v4h;
typedef __attribute__((ext_vector_type(8)))  float    v8f;
typedef __attribute__((ext_vector_type(4)))  float    v4f;
typedef __attribute__((ext_vector_type(4)))  unsigned int v4u;
typedef __attribute__((ext_vector_type(4)))  int      v4i;
typedef __attribute__((ext_vector_type(8)))  int      v8i;

#define WMMA_F16(a, b, c) \
  __builtin_amdgcn_wmma_f32_16x16x32_f16(false, (a), false, (b), (short)0, (c), false, false)

// Problem dimensions (from setup_inputs: B=4, S=1024, D=1024, H=16, FF=4096)
#define BB   4
#define SS   1024
#define DD   1024
#define HH   16
#define DHD  64
#define FF   4096
#define MROWS (BB * SS)   // 4096

// ---------------------------------------------------------------------------
// Tensor Data Mover: 2D tile (tile1 rows x tile0 elems, 2-byte elements) from
// global (row pitch stride0 elems) into LDS at byte offset lds_off, rows
// packed contiguously. Descriptor per CDNA5 ISA.
// ---------------------------------------------------------------------------
static __device__ __forceinline__ void tdm_load_2d_f16(
    unsigned int lds_off, const void* gsrc,
    unsigned int tensor_d0, unsigned int tensor_d1,
    unsigned int tile0, unsigned int tile1, unsigned int stride0) {
  const unsigned long long ga = (unsigned long long)(uintptr_t)gsrc;

  v4u g0;
  g0[0] = 1u;                                      // count=1, user descriptor
  g0[1] = lds_off;                                 // lds_addr (bytes)
  g0[2] = (unsigned int)ga;                        // global_addr[31:0]
  g0[3] = (unsigned int)((ga >> 32) & 0x01FFFFFFu) // global_addr[56:32]
          | 0x80000000u;                           // type=2 ("image") bits[127:126]

  v8i g1;
  g1[0] = (int)(1u << 16);                         // data_size=1 (2 bytes), no mask
  g1[1] = (int)((tensor_d0 & 0xFFFFu) << 16);      // tensor_dim0[15:0] -> bits[63:48]
  g1[2] = (int)((tensor_d0 >> 16) | ((tensor_d1 & 0xFFFFu) << 16)); // td0 hi | td1 lo
  g1[3] = (int)((tensor_d1 >> 16) | (tile0 << 16));                 // td1 hi | tile_dim0
  g1[4] = (int)(tile1 & 0xFFFFu);                  // tile_dim1 (tile_dim2 = 0)
  g1[5] = (int)stride0;                            // tensor_dim0_stride[31:0]
  g1[6] = 0;                                       // stride0[47:32]=0 | dim1_stride lo=0
  g1[7] = 0;

  v4i gz = {0, 0, 0, 0};
#if __clang_major__ >= 23
  v8i gz8 = {0, 0, 0, 0, 0, 0, 0, 0};
  __builtin_amdgcn_tensor_load_to_lds(g0, g1, gz, gz, gz8, 0);
#else
  __builtin_amdgcn_tensor_load_to_lds(g0, g1, gz, gz, 0);
#endif
}

// ---------------------------------------------------------------------------
// Operand loaders matching the CDNA5 WMMA VGPR layouts (CDNA5 ISA)
//
// A (16x32 f16, M x K):  lane m = lane%16 holds K = {0..7,16..23} (lanes 0-15)
//                        or K = {8..15,24..31} (lanes 16-31)  -> two 16B loads
// B (32x16 f16, K x N):  lane l holds col N=l%16, 16 K in A's order -> 16 strided 2B loads
// C/D (16x16 f32):       VGPR r, lane l -> row r + 8*(l/16), col l%16
// ---------------------------------------------------------------------------
static __device__ __forceinline__ v16h pack_a(v8h lo, v8h hi) {
  v16h a;
#pragma unroll
  for (int i = 0; i < 8; ++i) { a[i] = lo[i]; a[8 + i] = hi[i]; }
  return a;
}

// ---------------------------------------------------------------------------
// fp32 -> fp16 weight conversion (one-time per launch, streaming)
// ---------------------------------------------------------------------------
__global__ void cvt_f32_f16_kernel(const float* __restrict__ in,
                                   _Float16* __restrict__ out, int n) {
  int i = blockIdx.x * 256 + threadIdx.x;
  if (i < n) out[i] = (_Float16)in[i];
}

// ---------------------------------------------------------------------------
// Fused LayerNorm + block-floating-point quant (block=16, man_bits=4) -> f16
// ---------------------------------------------------------------------------
__global__ __launch_bounds__(256) void ln_bfp_kernel(const float* __restrict__ x,
                                                     const float* __restrict__ g,
                                                     const float* __restrict__ bsh,
                                                     _Float16* __restrict__ out) {
  const int row = blockIdx.x;
  const int tid = threadIdx.x;
  const float* xr = x + (size_t)row * DD;

  v4f v = *(const v4f*)(xr + tid * 4);
  float s  = v.x + v.y + v.z + v.w;
  float s2 = v.x * v.x + v.y * v.y + v.z * v.z + v.w * v.w;
#pragma unroll
  for (int m = 1; m < 32; m <<= 1) {
    s  += __shfl_xor(s,  m, 32);
    s2 += __shfl_xor(s2, m, 32);
  }
  __shared__ float ssum[8], ssq[8];
  const int wid = tid >> 5;
  if ((tid & 31) == 0) { ssum[wid] = s; ssq[wid] = s2; }
  __syncthreads();
  s = 0.f; s2 = 0.f;
#pragma unroll
  for (int i = 0; i < 8; ++i) { s += ssum[i]; s2 += ssq[i]; }
  const float mu   = s * (1.0f / DD);
  const float var  = s2 * (1.0f / DD) - mu * mu;
  const float rstd = rsqrtf(var + 1e-6f);

  float y[4];
  float amax = 0.f;
#pragma unroll
  for (int i = 0; i < 4; ++i) {
    const int c = tid * 4 + i;
    const float t = (((const float*)&v)[i] - mu) * rstd * g[c] + bsh[c];
    y[i] = t;
    amax = fmaxf(amax, fabsf(t));
  }
  amax = fmaxf(amax, __shfl_xor(amax, 1, 32));
  amax = fmaxf(amax, __shfl_xor(amax, 2, 32));

  const float e     = floorf(log2f(fmaxf(amax, 1e-30f)));
  const float scale = exp2f(e - 3.0f);           // man_bits-1 = 3
  const float inv   = 1.0f / scale;
  v4h q;
#pragma unroll
  for (int i = 0; i < 4; ++i) {
    float t = fminf(fmaxf(roundf(y[i] * inv), -7.f), 7.f) * scale;
    if (!(amax > 0.f)) t = y[i];
    q[i] = (_Float16)t;
  }
  *(v4h*)(out + (size_t)row * DD + tid * 4) = q;
}

// ---------------------------------------------------------------------------
// Tiled WMMA GEMM with TDM double-buffered LDS staging.
// C[M,N](f32 accum) = A[M,K](f16) * B[K,N](f16)
// 256 threads = 8 waves in 2x4; block tile 32(M) x 256(N), k-step 32.
// Per k-step, TDM DMAs A tile (32x32) and B tile (32x256) into LDS
// (TENSORcnt-tracked), waves consume via ds loads (16B for A, 2B for B) in WMMA layouts.
// EPI: 0 = f16 out; 1 = +bias, exact GELU, f16; 2 = +bias +residual, f32
// ---------------------------------------------------------------------------
template <int EPI>
__global__ __launch_bounds__(256) void wmma_gemm_kernel(
    const _Float16* __restrict__ A, const _Float16* __restrict__ Bm,
    const float* __restrict__ bias, const float* __restrict__ res,
    void* __restrict__ out, int M, int N, int K) {
  const int lane = threadIdx.x & 31;
  const int wid  = threadIdx.x >> 5;
  const int wm = wid >> 2, wn = wid & 3;
  const int mb = blockIdx.y * 32;             // block row base
  const int n0 = blockIdx.x * 256 + wn * 64;  // wave col base
  const int m0 = mb + wm * 16;                // wave row base

  __shared__ _Float16 bstage[2][32 * 256];    // 2 x 16 KB
  __shared__ _Float16 astage[2][32 * 32];     // 2 x  2 KB

  v8f c[4] = {};

  // prime buffer 0
  if (wid == 0) {
    tdm_load_2d_f16((unsigned int)(uintptr_t)&bstage[0][0],
                    Bm + blockIdx.x * 256, (unsigned int)N, (unsigned int)K,
                    256u, 32u, (unsigned int)N);
    tdm_load_2d_f16((unsigned int)(uintptr_t)&astage[0][0],
                    A + (size_t)mb * K, (unsigned int)K, (unsigned int)M,
                    32u, 32u, (unsigned int)K);
  }

  int cur = 0;
  for (int k0 = 0; k0 < K; k0 += 32) {
    const int nxt = cur ^ 1;
    const bool more = (k0 + 32 < K);
    if (wid == 0) {
      if (more) {
        tdm_load_2d_f16((unsigned int)(uintptr_t)&bstage[nxt][0],
                        Bm + (size_t)(k0 + 32) * N + blockIdx.x * 256,
                        (unsigned int)N, (unsigned int)K, 256u, 32u,
                        (unsigned int)N);
        tdm_load_2d_f16((unsigned int)(uintptr_t)&astage[nxt][0],
                        A + (size_t)mb * K + (k0 + 32),
                        (unsigned int)K, (unsigned int)M, 32u, 32u,
                        (unsigned int)K);
        // current pair done when only the freshly-issued pair is in flight
        __builtin_amdgcn_s_wait_tensorcnt(2);
      } else {
        __builtin_amdgcn_s_wait_tensorcnt(0);
      }
    }
    __syncthreads();  // publish LDS tiles to all 8 waves

    // A operand (16x32) for this wave's row strip, from LDS
    const _Float16* as = &astage[cur][0];
    const int am = wm * 16 + (lane & 15);
    const int kb = (lane < 16) ? 0 : 8;
    const v8h alo = *(const v8h*)(as + am * 32 + kb);
    const v8h ahi = *(const v8h*)(as + am * 32 + kb + 16);
    const v16h a = pack_a(alo, ahi);

    // B operands (32x16 x4) : lane l holds col N=l%16, K in A's table order
    const _Float16* bs = &bstage[cur][0] + wn * 64 + (lane & 15);
    v16h b0, b1, b2, b3;
#pragma unroll
    for (int s = 0; s < 16; ++s) {
      const int kk = (s & 7) + ((lane >> 4) << 3) + ((s >> 3) << 4);
      b0[s] = bs[kk * 256];
      b1[s] = bs[kk * 256 + 16];
      b2[s] = bs[kk * 256 + 32];
      b3[s] = bs[kk * 256 + 48];
    }

    c[0] = WMMA_F16(a, b0, c[0]);
    c[1] = WMMA_F16(a, b1, c[1]);
    c[2] = WMMA_F16(a, b2, c[2]);
    c[3] = WMMA_F16(a, b3, c[3]);

    __syncthreads();  // all waves done reading before TDM overwrites `cur`
    cur = nxt;
  }

  const int rbase = m0 + ((lane >> 4) << 3);
  const int cl    = lane & 15;
#pragma unroll
  for (int j = 0; j < 4; ++j) {
    const int col = n0 + j * 16 + cl;
#pragma unroll
    for (int r = 0; r < 8; ++r) {
      const size_t idx = (size_t)(rbase + r) * N + col;
      const float v = c[j][r];
      if (EPI == 0) {
        ((_Float16*)out)[idx] = (_Float16)v;
      } else if (EPI == 1) {
        const float t = v + bias[col];
        const float gl = 0.5f * t * (1.0f + erff(t * 0.70710678118654752f));
        ((_Float16*)out)[idx] = (_Float16)gl;
      } else {
        ((float*)out)[idx] = res[idx] + v + bias[col];
      }
    }
  }
}

// ---------------------------------------------------------------------------
// Pre-transpose K from qkv[b,s, D + h*64 + d] into kT[b,h,d,s]
// ---------------------------------------------------------------------------
__global__ __launch_bounds__(256) void kT_kernel(const _Float16* __restrict__ qkv,
                                                 _Float16* __restrict__ kT) {
  const int i  = blockIdx.x * 256 + threadIdx.x;  // B*H*64*S total
  const int s  = i & (SS - 1);
  const int d  = (i >> 10) & 63;
  const int bh = i >> 16;
  const int b  = bh >> 4, h = bh & 15;
  kT[i] = qkv[((size_t)(b * SS + s)) * (3 * DD) + DD + h * DHD + d];
}

// ---------------------------------------------------------------------------
// Flash-style attention, all-WMMA. 8 waves/block, each wave owns a 16-query
// tile of one (b,h); online softmax in f32; P relaid C/D->A layout via LDS.
// ---------------------------------------------------------------------------
__global__ __launch_bounds__(256) void attn_kernel(const _Float16* __restrict__ qkv,
                                                   const _Float16* __restrict__ kT,
                                                   _Float16* __restrict__ o) {
  const int lane = threadIdx.x & 31;
  const int wid  = threadIdx.x >> 5;
  const int h = blockIdx.y, b = blockIdx.z;
  const int s0 = (blockIdx.x * 8 + wid) * 16;
  const float scl = 0.125f;  // 64^-0.5

  __shared__ _Float16 pbuf[8][16 * 32];
  _Float16* pb = pbuf[wid];

  // Q tile in A layout, d-chunks {0..31, 32..63}
  v16h qa[2];
  {
    const int m  = lane & 15;
    const int kb = (lane < 16) ? 0 : 8;
    const _Float16* base = qkv + ((size_t)(b * SS + s0 + m)) * (3 * DD) + h * DHD;
#pragma unroll
    for (int dc = 0; dc < 2; ++dc) {
      const _Float16* p = base + dc * 32 + kb;
      qa[dc] = pack_a(*(const v8h*)p, *(const v8h*)(p + 16));
    }
  }

  float run_m[8], run_l[8];
  v8f oacc[4] = {};
#pragma unroll
  for (int r = 0; r < 8; ++r) { run_m[r] = -3.0e38f; run_l[r] = 0.f; }

  const _Float16* kbase = kT + ((size_t)(b * HH + h) * DHD) * SS;

  for (int t0 = 0; t0 < SS; t0 += 32) {
    // ---- scores: S[16x32] = Q[16x64] * K^T[64x32], 4 chained WMMAs ----
    v8f sc0 = {}, sc1 = {};
#pragma unroll
    for (int dc = 0; dc < 2; ++dc) {
      const _Float16* kp = kbase + (size_t)(dc * 32) * SS + t0 + (lane & 15);
      v16h kt0, kt1;
#pragma unroll
      for (int s = 0; s < 16; ++s) {
        const int kk = (s & 7) + ((lane >> 4) << 3) + ((s >> 3) << 4);
        kt0[s] = kp[(size_t)kk * SS];
        kt1[s] = kp[(size_t)kk * SS + 16];
      }
      sc0 = WMMA_F16(qa[dc], kt0, sc0);
      sc1 = WMMA_F16(qa[dc], kt1, sc1);
    }

    // ---- online softmax (per-row across a 16-lane group) ----
#pragma unroll
    for (int r = 0; r < 8; ++r) {
      const float a0 = sc0[r] * scl;
      const float a1 = sc1[r] * scl;
      float mx = fmaxf(a0, a1);
      mx = fmaxf(mx, __shfl_xor(mx, 1, 32));
      mx = fmaxf(mx, __shfl_xor(mx, 2, 32));
      mx = fmaxf(mx, __shfl_xor(mx, 4, 32));
      mx = fmaxf(mx, __shfl_xor(mx, 8, 32));
      const float nm   = fmaxf(run_m[r], mx);
      const float corr = __expf(run_m[r] - nm);
      run_m[r] = nm;
      const float p0 = __expf(a0 - nm);
      const float p1 = __expf(a1 - nm);
      float rs = p0 + p1;
      rs += __shfl_xor(rs, 1, 32);
      rs += __shfl_xor(rs, 2, 32);
      rs += __shfl_xor(rs, 4, 32);
      rs += __shfl_xor(rs, 8, 32);
      run_l[r] = run_l[r] * corr + rs;
#pragma unroll
      for (int j = 0; j < 4; ++j) oacc[j][r] *= corr;
      const int row = r + ((lane >> 4) << 3);
      pb[row * 32 + (lane & 15)]      = (_Float16)p0;
      pb[row * 32 + 16 + (lane & 15)] = (_Float16)p1;
    }
    __syncthreads();

    // ---- reload P in A layout ----
    const int m  = lane & 15;
    const int kb = (lane < 16) ? 0 : 8;
    const v16h pa = pack_a(*(const v8h*)(pb + m * 32 + kb),
                           *(const v8h*)(pb + m * 32 + kb + 16));

    // ---- O[16x64] += P[16x32] * V[32x64], 4 WMMAs ----
    const _Float16* vrow =
        qkv + ((size_t)(b * SS + t0)) * (3 * DD) + 2 * DD + h * DHD + (lane & 15);
#pragma unroll
    for (int j = 0; j < 4; ++j) {
      v16h vb;
#pragma unroll
      for (int s = 0; s < 16; ++s) {
        const int kk = (s & 7) + ((lane >> 4) << 3) + ((s >> 3) << 4);
        vb[s] = vrow[(size_t)kk * (3 * DD) + j * 16];
      }
      oacc[j] = WMMA_F16(pa, vb, oacc[j]);
    }
    __syncthreads();  // protect pbuf WAR for next iteration
  }

  // ---- normalize and write o[b,s, h*64 + d] as f16 ----
#pragma unroll
  for (int r = 0; r < 8; ++r) {
    const int row = s0 + r + ((lane >> 4) << 3);
    const float inv = 1.0f / run_l[r];
#pragma unroll
    for (int j = 0; j < 4; ++j) {
      o[((size_t)(b * SS + row)) * DD + h * DHD + j * 16 + (lane & 15)] =
          (_Float16)(oacc[j][r] * inv);
    }
  }
}

// ---------------------------------------------------------------------------
// Host orchestration
// ---------------------------------------------------------------------------
extern "C" void kernel_launch(void* const* d_in, const int* in_sizes, int n_in,
                              void* d_out, int out_size, void* d_ws, size_t ws_size,
                              hipStream_t stream) {
  (void)in_sizes; (void)n_in; (void)out_size; (void)ws_size;
  const float* x      = (const float*)d_in[0];
  const float* ln1_g  = (const float*)d_in[1];
  const float* ln1_b  = (const float*)d_in[2];
  const float* ln2_g  = (const float*)d_in[3];
  const float* ln2_b  = (const float*)d_in[4];
  const float* w_qkv  = (const float*)d_in[5];
  const float* w_proj = (const float*)d_in[6];
  const float* b_proj = (const float*)d_in[7];
  const float* w_fc1  = (const float*)d_in[8];
  const float* b_fc1  = (const float*)d_in[9];
  const float* w_fc2  = (const float*)d_in[10];
  const float* b_fc2  = (const float*)d_in[11];

  // workspace carve-up (all 256B-aligned sizes)
  char* ws = (char*)d_ws;
  _Float16* wqkv_h  = (_Float16*)(ws);                               // 6 MiB
  _Float16* wproj_h = (_Float16*)(ws + (size_t)6 * 1024 * 1024);     // 2 MiB
  _Float16* wfc1_h  = (_Float16*)(ws + (size_t)8 * 1024 * 1024);     // 8 MiB
  _Float16* wfc2_h  = (_Float16*)(ws + (size_t)16 * 1024 * 1024);    // 8 MiB
  _Float16* h1      = (_Float16*)(ws + (size_t)24 * 1024 * 1024);    // 8 MiB
  _Float16* qkv     = (_Float16*)(ws + (size_t)32 * 1024 * 1024);    // 24 MiB
  _Float16* kT      = (_Float16*)(ws + (size_t)56 * 1024 * 1024);    // 8 MiB
  _Float16* ob      = (_Float16*)(ws + (size_t)64 * 1024 * 1024);    // 8 MiB
  float*    x1      = (float*)   (ws + (size_t)72 * 1024 * 1024);    // 16 MiB
  _Float16* h2      = (_Float16*)(ws + (size_t)88 * 1024 * 1024);    // 8 MiB
  _Float16* mbuf    = (_Float16*)(ws + (size_t)96 * 1024 * 1024);    // 32 MiB

  // 1. convert fp32 weights to f16 once per launch
  cvt_f32_f16_kernel<<<(DD * 3 * DD + 255) / 256, 256, 0, stream>>>(w_qkv,  wqkv_h,  DD * 3 * DD);
  cvt_f32_f16_kernel<<<(DD * DD     + 255) / 256, 256, 0, stream>>>(w_proj, wproj_h, DD * DD);
  cvt_f32_f16_kernel<<<(DD * FF     + 255) / 256, 256, 0, stream>>>(w_fc1,  wfc1_h,  DD * FF);
  cvt_f32_f16_kernel<<<(FF * DD     + 255) / 256, 256, 0, stream>>>(w_fc2,  wfc2_h,  FF * DD);

  // 2. LN1 + BFP quant
  ln_bfp_kernel<<<MROWS, 256, 0, stream>>>(x, ln1_g, ln1_b, h1);

  // 3. qkv = h1 @ w_qkv    (4096 x 3072 x 1024), f16 out
  wmma_gemm_kernel<0><<<dim3(3 * DD / 256, MROWS / 32), 256, 0, stream>>>(
      h1, wqkv_h, nullptr, nullptr, qkv, MROWS, 3 * DD, DD);

  // 4. transpose K for the QK^T B-operand
  kT_kernel<<<(BB * HH * DHD * SS) / 256, 256, 0, stream>>>(qkv, kT);

  // 5. attention
  attn_kernel<<<dim3(SS / 16 / 8, HH, BB), 256, 0, stream>>>(qkv, kT, ob);

  // 6. x1 = x + o @ w_proj + b_proj   (f32)
  wmma_gemm_kernel<2><<<dim3(DD / 256, MROWS / 32), 256, 0, stream>>>(
      ob, wproj_h, b_proj, x, x1, MROWS, DD, DD);

  // 7. LN2 + BFP quant
  ln_bfp_kernel<<<MROWS, 256, 0, stream>>>(x1, ln2_g, ln2_b, h2);

  // 8. m = gelu(h2 @ w_fc1 + b_fc1), f16 out
  wmma_gemm_kernel<1><<<dim3(FF / 256, MROWS / 32), 256, 0, stream>>>(
      h2, wfc1_h, b_fc1, nullptr, mbuf, MROWS, FF, DD);

  // 9. out = x1 + m @ w_fc2 + b_fc2   (f32)
  wmma_gemm_kernel<2><<<dim3(DD / 256, MROWS / 32), 256, 0, stream>>>(
      mbuf, wfc2_h, b_fc2, x1, d_out, MROWS, DD, FF);
}
